// UnitGCN_55138790146457
// MI455X (gfx1250) — matrix operand order repaired and checked by reading, not compiled, not run on hardware
//
#include <hip/hip_runtime.h>
#include <hip/hip_bf16.h>
#include <cstdint>

// ---------------- problem constants ----------------
#define NN   64
#define CIN  128
#define TT   128
#define VV   25
#define COUT 256
#define KK   3
#define MID  32
#define JJ   96          // K*MID
#define SS   3200        // T*V
#define EPSB 1e-5f

typedef __attribute__((ext_vector_type(16))) __bf16    bf16x16;
typedef __attribute__((ext_vector_type(8)))  float     f32x8;
typedef __attribute__((ext_vector_type(4)))  uint32_t  u32x4;

union FragAB { u32x4 q[2]; uint32_t u[8]; bf16x16 v; };

__device__ __forceinline__ uint16_t f2bf(float f) {
  union { float f; uint32_t u; } c; c.f = f;
  uint32_t u = c.u;
  u += 0x7FFFu + ((u >> 16) & 1u);       // RNE truncate to bf16
  return (uint16_t)(u >> 16);
}

// ---- swizzled-tile index helpers (one operand tile = 512 bf16, per-lane 16) ----
// A operand (16M x 32K, lane-half selects K 0-15/16-31, 2 per VGPR): element (m,k) -> tile index
__device__ __forceinline__ int aidx(int m, int k) {
  int lane = (m & 15) + (((k >> 4) & 1) << 4);
  int r    = (k >> 1) & 7;
  return lane * 16 + (r << 1) + (k & 1);
}
// B operand (32K x 16N, lane-half selects K 0-15/16-31, 2 per VGPR): element (k,s)
__device__ __forceinline__ int bidx(int k, int s) {
  int lane = (((k >> 4) & 1) << 4) + (s & 15);
  int r    = (k >> 1) & 7;
  return lane * 16 + (r << 1) + (k & 1);
}

// ---------------- kernel Z: zero a buffer (pad lanes of pre_sw) ----------------
__global__ void k_zero(u32x4* __restrict__ p, int n4) {
  int i = blockIdx.x * 256 + threadIdx.x;
  if (i < n4) { u32x4 z = {0u, 0u, 0u, 0u}; p[i] = z; }
}

// ---------------- kernel 0: fold BN coeffs, swizzle weights to bf16 tiles ------
__global__ void k_coeffs(const float* __restrict__ w_pre, const float* __restrict__ w_down,
                         const float* __restrict__ w_post,
                         const float* __restrict__ b_pre, const float* __restrict__ g_pre,
                         const float* __restrict__ bb_pre, const float* __restrict__ m_pre,
                         const float* __restrict__ v_pre,
                         const float* __restrict__ b_post, const float* __restrict__ b_down,
                         const float* __restrict__ g_down, const float* __restrict__ bb_down,
                         const float* __restrict__ m_down, const float* __restrict__ v_down,
                         const float* __restrict__ g_bn, const float* __restrict__ bb_bn,
                         const float* __restrict__ m_bn, const float* __restrict__ v_bn,
                         uint16_t* __restrict__ w_pre_sw, uint16_t* __restrict__ w_down_sw,
                         uint16_t* __restrict__ w_post_sw,
                         float* __restrict__ pre_scale, float* __restrict__ pre_shift,
                         float* __restrict__ sbn, float* __restrict__ sdn,
                         float* __restrict__ fshift) {
  int idx = blockIdx.x * 256 + threadIdx.x;
  if (idx < JJ * CIN) {                    // w_pre: 6 mt x 4 kb tiles
    int o = idx / CIN, c = idx % CIN;
    w_pre_sw[((o >> 4) * 4 + (c >> 5)) * 512 + aidx(o & 15, c & 31)] = f2bf(w_pre[idx]);
    return;
  }
  idx -= JJ * CIN;
  if (idx < COUT * CIN) {                  // w_down: 16 mt x 4 kb tiles
    int o = idx / CIN, c = idx % CIN;
    w_down_sw[((o >> 4) * 4 + (c >> 5)) * 512 + aidx(o & 15, c & 31)] = f2bf(w_down[idx]);
    return;
  }
  idx -= COUT * CIN;
  if (idx < COUT * JJ) {                   // w_post: 16 mt x 3 kb tiles
    int o = idx / JJ, c = idx % JJ;
    w_post_sw[((o >> 4) * 3 + (c >> 5)) * 512 + aidx(o & 15, c & 31)] = f2bf(w_post[idx]);
    return;
  }
  idx -= COUT * JJ;
  if (idx < JJ) {
    float s = g_pre[idx] * rsqrtf(v_pre[idx] + EPSB);
    pre_scale[idx] = s;
    pre_shift[idx] = s * (b_pre[idx] - m_pre[idx]) + bb_pre[idx];
    return;
  }
  idx -= JJ;
  if (idx < COUT) {
    float s1 = g_bn[idx]   * rsqrtf(v_bn[idx]   + EPSB);
    float s2 = g_down[idx] * rsqrtf(v_down[idx] + EPSB);
    sbn[idx] = s1;
    sdn[idx] = s2;
    fshift[idx] = s1 * (b_post[idx] - m_bn[idx])   + bb_bn[idx]
                + s2 * (b_down[idx] - m_down[idx]) + bb_down[idx];
  }
}

// ---------------- kernel 1: x -> swizzled bf16 B-tiles + time mean -------------
__global__ void k_reduce(const float* __restrict__ x, uint16_t* __restrict__ x_sw,
                         float* __restrict__ xbar) {
  __shared__ uint16_t stage[25 * 512];     // 25 s-tiles (400 s) x 32 k rows
  __shared__ float    sums[32 * VV];
  const int n = blockIdx.x >> 2, kb = blockIdx.x & 3;
  const int tid = threadIdx.x;
  for (int i = tid; i < 32 * VV; i += 256) sums[i] = 0.f;
  __syncthreads();
  const float* xp = x + ((size_t)n * CIN + kb * 32) * SS;
  uint16_t*    op = x_sw + (((size_t)n * 4 + kb) * 200) * 512;
  for (int chunk = 0; chunk < 8; ++chunk) {          // 8 x 400 = 3200 s
    const int s0 = chunk * 400;                      // multiple of both 16 and 25
    for (int i = tid; i < 32 * 400; i += 256) {
      int kl = i / 400, sl = i % 400;
      float v = xp[(size_t)kl * SS + s0 + sl];
      atomicAdd(&sums[kl * VV + (sl % VV)], v);
      stage[(sl >> 4) * 512 + bidx(kl, sl)] = f2bf(v);
    }
    __syncthreads();
    const uint32_t* s32 = (const uint32_t*)stage;
    uint32_t*       o32 = (uint32_t*)(op + (size_t)chunk * 25 * 512);
    for (int i = tid; i < 25 * 256; i += 256) o32[i] = s32[i];
    __syncthreads();
  }
  for (int i = tid; i < 32 * VV; i += 256) {
    int kl = i / VV, v = i % VV;
    xbar[((size_t)n * CIN + kb * 32 + kl) * VV + v] = sums[i] * (1.f / (float)TT);
  }
}

// ---------------- kernel 2: per-(n,k) graph construction ----------------
__global__ void k_graph(const float* __restrict__ xbar,
                        const float* __restrict__ w1, const float* __restrict__ b1,
                        const float* __restrict__ w2, const float* __restrict__ b2,
                        const float* __restrict__ A,
                        const float* __restrict__ alpha, const float* __restrict__ beta,
                        float* __restrict__ Agg, float* __restrict__ gg_out) {
  __shared__ float xs[CIN * VV];
  __shared__ float x1s[MID * VV];
  __shared__ float x2s[MID * VV];
  __shared__ float sc[VV * VV];
  const int n = blockIdx.x / KK, k = blockIdx.x % KK;
  const int tid = threadIdx.x;
  const float* xb = xbar + (size_t)n * CIN * VV;
  for (int i = tid; i < CIN * VV; i += 256) xs[i] = xb[i];
  __syncthreads();
  for (int i = tid; i < MID * VV; i += 256) {    // x1/x2 = conv(mean_t(x))
    int jl = i / VV, v = i % VV;
    int j = k * MID + jl;
    float a1 = b1[j], a2 = b2[j];
    const float* w1r = w1 + (size_t)j * CIN;
    const float* w2r = w2 + (size_t)j * CIN;
    for (int c = 0; c < CIN; ++c) {
      float xv = xs[c * VV + v];
      a1 += w1r[c] * xv;
      a2 += w2r[c] * xv;
    }
    x1s[i] = a1; x2s[i] = a2;
  }
  __syncthreads();
  for (int i = tid; i < VV * VV; i += 256) {     // scores
    int v = i / VV, w = i % VV;
    float acc = 0.f;
    for (int jl = 0; jl < MID; ++jl) acc += x1s[jl * VV + v] * x2s[jl * VV + w];
    sc[i] = acc;
  }
  __syncthreads();
  const float beta0 = beta[0];
  if (tid < VV) {                                // column softmax over v
    int w = tid;
    float mx = -3.4e38f;
    for (int v = 0; v < VV; ++v) mx = fmaxf(mx, sc[v * VV + w]);
    float sm = 0.f;
    for (int v = 0; v < VV; ++v) sm += __expf(sc[v * VV + w] - mx);
    float inv = beta0 / sm;
    for (int v = 0; v < VV; ++v) sc[v * VV + w] = __expf(sc[v * VV + w] - mx) * inv;
  }
  __syncthreads();
  const float alpha0 = alpha[0];
  const float* Ak = A + (size_t)k * VV * VV;
  const size_t obase = ((size_t)n * JJ + k * MID) * (VV * VV);
  for (int i = tid; i < MID * VV * VV; i += 256) {
    int jl = i / (VV * VV), r = i % (VV * VV);
    int v = r / VV, w = r % VV;
    float inter = tanhf(x1s[jl * VV + v] - x2s[jl * VV + w]) * alpha0;
    float g = inter + sc[r];
    gg_out[obase + i] = g;
    Agg[obase + i]    = g + Ak[r];
  }
}

// -------- kernel 3: pre = relu(bn(w_pre @ x)), 4 s-tiles/wave, A reuse ---------
// 4 independent accumulator chains fully cover the bf16 WMMA->WMMA RAW hazard
// window (1 NOP + 4 co-exec slots) with useful work.
__global__ void k_pre(const uint16_t* __restrict__ x_sw, const uint16_t* __restrict__ w_sw,
                      const float* __restrict__ scale, const float* __restrict__ shift,
                      uint16_t* __restrict__ pre_sw) {
  const int wid  = blockIdx.x * 8 + (threadIdx.x >> 5);
  const int lane = threadIdx.x & 31;
  const int n = wid / 300;
  const int rem = wid % 300;
  const int mt = rem / 50, st = (rem % 50) * 4;     // tile quad
  const int col = lane & 15;
  const int hi8 = (lane >> 4) << 3;
  const uint16_t* wp = w_sw + (size_t)mt * 4 * 512;
  const uint16_t* xp = x_sw + (((size_t)n * 4) * 200 + st) * 512;
  f32x8 acc0 = {}, acc1 = {}, acc2 = {}, acc3 = {};
#pragma unroll
  for (int kb = 0; kb < 4; ++kb) {
    if (kb + 1 < 4) __builtin_prefetch(xp + ((size_t)kb + 1) * 200 * 512, 0, 1);
    FragAB a, b0, b1, b2, b3;
    const u32x4* aq = (const u32x4*)(wp + kb * 512) + lane * 2;
    a.q[0] = aq[0]; a.q[1] = aq[1];
    const u32x4* bq = (const u32x4*)(xp + (size_t)kb * 200 * 512) + lane * 2;
    b0.q[0] = bq[0];   b0.q[1] = bq[1];
    b1.q[0] = bq[64];  b1.q[1] = bq[65];            // +1 s-tile (512 bf16)
    b2.q[0] = bq[128]; b2.q[1] = bq[129];
    b3.q[0] = bq[192]; b3.q[1] = bq[193];
    acc0 = __builtin_amdgcn_wmma_f32_16x16x32_bf16(false, a.v, false, b0.v,
                                                   (short)0, acc0, false, false);
    acc1 = __builtin_amdgcn_wmma_f32_16x16x32_bf16(false, a.v, false, b1.v,
                                                   (short)0, acc1, false, false);
    acc2 = __builtin_amdgcn_wmma_f32_16x16x32_bf16(false, a.v, false, b2.v,
                                                   (short)0, acc2, false, false);
    acc3 = __builtin_amdgcn_wmma_f32_16x16x32_bf16(false, a.v, false, b3.v,
                                                   (short)0, acc3, false, false);
  }
  // epilogue: scatter into A-swizzled tiles (n, j, ttile) for the aggregation GEMM
#pragma unroll
  for (int u = 0; u < 4; ++u) {
    const int s = (st + u) * 16 + col;
    const int t = s / VV, v = s - t * VV;
    const int ttile = t >> 4;
    const int idx = aidx(t & 15, v);
    uint16_t* op = pre_sw + (((size_t)n * JJ) * 8 + ttile) * 512 + idx;
    f32x8 acc = (u == 0) ? acc0 : (u == 1) ? acc1 : (u == 2) ? acc2 : acc3;
#pragma unroll
    for (int r = 0; r < 8; ++r) {
      int j = mt * 16 + r + hi8;
      float val = acc[r] * scale[j] + shift[j];
      op[(size_t)j * 8 * 512] = f2bf(fmaxf(val, 0.f));
    }
  }
}

// ------- kernel 4: graph aggregation (128x25 @ 25x25, padded WMMA tiles) -------
__global__ void k_agg(const uint16_t* __restrict__ pre_sw, const float* __restrict__ Agg,
                      uint16_t* __restrict__ y_sw) {
  const int b    = blockIdx.x;                 // n*96 + j
  const int n    = b / JJ, j = b % JJ;
  const int wv   = threadIdx.x >> 5;           // 4 waves, 2 t-tiles each
  const int lane = threadIdx.x & 31;
  const int col  = lane & 15;
  const int hi8  = (lane >> 4) << 3;
  const uint16_t* pp = pre_sw + (size_t)b * 8 * 512;
  const float*    ag = Agg    + (size_t)b * (VV * VV);
  // build both B fragments (w-tiles 0 and 1) once; K = v, zero-padded to 32
  FragAB b0, b1;
#pragma unroll
  for (int r = 0; r < 8; ++r) {
    int kb = (((lane >> 4) & 1) << 4) | (r << 1);
    int w0 = col, w1 = 16 + col;
    float a00 = (kb     < VV) ? ag[kb * VV + w0]       : 0.f;
    float a01 = (kb + 1 < VV) ? ag[(kb + 1) * VV + w0] : 0.f;
    b0.u[r] = (uint32_t)f2bf(a00) | ((uint32_t)f2bf(a01) << 16);
    float a10 = (kb     < VV && w1 < VV) ? ag[kb * VV + w1]       : 0.f;
    float a11 = (kb + 1 < VV && w1 < VV) ? ag[(kb + 1) * VV + w1] : 0.f;
    b1.u[r] = (uint32_t)f2bf(a10) | ((uint32_t)f2bf(a11) << 16);
  }
  const int kb3 = j >> 5, kl = j & 31;
  const int koff = ((((kl >> 4) & 1) << 4) << 4) + (((kl >> 1) & 7) << 1) + (kl & 1);
  uint16_t* yb = y_sw + ((size_t)n * 3 + kb3) * 200 * 512;
  for (int tt = wv * 2; tt < wv * 2 + 2; ++tt) {
    FragAB a;
    const u32x4* aq = (const u32x4*)(pp + tt * 512) + lane * 2;
    a.q[0] = aq[0]; a.q[1] = aq[1];
    f32x8 acc0 = {}, acc1 = {};
    acc0 = __builtin_amdgcn_wmma_f32_16x16x32_bf16(false, a.v, false, b0.v,
                                                   (short)0, acc0, false, false);
    acc1 = __builtin_amdgcn_wmma_f32_16x16x32_bf16(false, a.v, false, b1.v,
                                                   (short)0, acc1, false, false);
    // scatter into B-swizzled tiles (k-dim = j) for the final GEMM
#pragma unroll
    for (int r = 0; r < 8; ++r) {
      int t = tt * 16 + r + hi8;
      int s0 = t * VV + col;
      yb[(size_t)(s0 >> 4) * 512 + koff + ((s0 & 15) << 4)] = f2bf(acc0[r]);
    }
    if (col < VV - 16) {
#pragma unroll
      for (int r = 0; r < 8; ++r) {
        int t = tt * 16 + r + hi8;
        int s1 = t * VV + 16 + col;
        yb[(size_t)(s1 >> 4) * 512 + koff + ((s1 & 15) << 4)] = f2bf(acc1[r]);
      }
    }
  }
}

// ---- kernel 5: out = relu(bn(w_post@yagg) + bn(w_down@x)), 2 s-tiles/wave -----
// Interleaved dual GEMM: 4 independent accumulator chains.
__global__ void k_final(const uint16_t* __restrict__ y_sw, const uint16_t* __restrict__ x_sw,
                        const uint16_t* __restrict__ w_post_sw, const uint16_t* __restrict__ w_down_sw,
                        const float* __restrict__ sbn, const float* __restrict__ sdn,
                        const float* __restrict__ fshift, float* __restrict__ out) {
  const int wid  = blockIdx.x * 8 + (threadIdx.x >> 5);
  const int lane = threadIdx.x & 31;
  const int n = wid / 1600;
  const int rem = wid % 1600;
  const int ot = rem / 100, st = (rem % 100) * 2;   // tile pair
  const int col = lane & 15;
  const int hi8 = (lane >> 4) << 3;
  const uint16_t* yp = y_sw + (((size_t)n * 3) * 200 + st) * 512;
  const uint16_t* xp = x_sw + (((size_t)n * 4) * 200 + st) * 512;
  const uint16_t* wpp = w_post_sw + (size_t)ot * 3 * 512;
  const uint16_t* wdp = w_down_sw + (size_t)ot * 4 * 512;
  f32x8 accp0 = {}, accp1 = {}, accd0 = {}, accd1 = {};
#pragma unroll
  for (int kb = 0; kb < 4; ++kb) {
    if (kb + 1 < 4) {
      __builtin_prefetch(xp + ((size_t)kb + 1) * 200 * 512, 0, 1);
      if (kb + 1 < 3) __builtin_prefetch(yp + ((size_t)kb + 1) * 200 * 512, 0, 1);
    }
    // down GEMM step (K = 128)
    {
      FragAB a, b0, b1;
      const u32x4* aq = (const u32x4*)(wdp + kb * 512) + lane * 2;
      a.q[0] = aq[0]; a.q[1] = aq[1];
      const u32x4* bq = (const u32x4*)(xp + (size_t)kb * 200 * 512) + lane * 2;
      b0.q[0] = bq[0];  b0.q[1] = bq[1];
      b1.q[0] = bq[64]; b1.q[1] = bq[65];
      accd0 = __builtin_amdgcn_wmma_f32_16x16x32_bf16(false, a.v, false, b0.v,
                                                      (short)0, accd0, false, false);
      accd1 = __builtin_amdgcn_wmma_f32_16x16x32_bf16(false, a.v, false, b1.v,
                                                      (short)0, accd1, false, false);
    }
    // post GEMM step (K = 96)
    if (kb < 3) {
      FragAB a, b0, b1;
      const u32x4* aq = (const u32x4*)(wpp + kb * 512) + lane * 2;
      a.q[0] = aq[0]; a.q[1] = aq[1];
      const u32x4* bq = (const u32x4*)(yp + (size_t)kb * 200 * 512) + lane * 2;
      b0.q[0] = bq[0];  b0.q[1] = bq[1];
      b1.q[0] = bq[64]; b1.q[1] = bq[65];
      accp0 = __builtin_amdgcn_wmma_f32_16x16x32_bf16(false, a.v, false, b0.v,
                                                      (short)0, accp0, false, false);
      accp1 = __builtin_amdgcn_wmma_f32_16x16x32_bf16(false, a.v, false, b1.v,
                                                      (short)0, accp1, false, false);
    }
  }
  float* op = out + (size_t)n * COUT * SS + st * 16 + col;
#pragma unroll
  for (int r = 0; r < 8; ++r) {
    int o = ot * 16 + r + hi8;
    float v0 = sbn[o] * accp0[r] + sdn[o] * accd0[r] + fshift[o];
    float v1 = sbn[o] * accp1[r] + sdn[o] * accd1[r] + fshift[o];
    op[(size_t)o * SS]      = fmaxf(v0, 0.f);
    op[(size_t)o * SS + 16] = fmaxf(v1, 0.f);
  }
}

// ---------------- host-side launch ----------------
extern "C" void kernel_launch(void* const* d_in, const int* in_sizes, int n_in,
                              void* d_out, int out_size, void* d_ws, size_t ws_size,
                              hipStream_t stream) {
  const float* x      = (const float*)d_in[0];
  const float* A      = (const float*)d_in[1];
  const float* alpha  = (const float*)d_in[2];
  const float* beta   = (const float*)d_in[3];
  const float* w_pre  = (const float*)d_in[4];
  const float* b_pre  = (const float*)d_in[5];
  const float* g_pre  = (const float*)d_in[6];
  const float* bb_pre = (const float*)d_in[7];
  const float* m_pre  = (const float*)d_in[8];
  const float* v_pre  = (const float*)d_in[9];
  const float* w1     = (const float*)d_in[10];
  const float* b1     = (const float*)d_in[11];
  const float* w2     = (const float*)d_in[12];
  const float* b2     = (const float*)d_in[13];
  const float* w_post = (const float*)d_in[14];
  const float* b_post = (const float*)d_in[15];
  const float* w_down = (const float*)d_in[16];
  const float* b_down = (const float*)d_in[17];
  const float* g_down = (const float*)d_in[18];
  const float* bb_down= (const float*)d_in[19];
  const float* m_down = (const float*)d_in[20];
  const float* v_down = (const float*)d_in[21];
  const float* g_bn   = (const float*)d_in[22];
  const float* bb_bn  = (const float*)d_in[23];
  const float* m_bn   = (const float*)d_in[24];
  const float* v_bn   = (const float*)d_in[25];

  float* out = (float*)d_out;
  float* gg  = out + (size_t)NN * COUT * SS;   // get_graph region

  char* ws = (char*)d_ws;
  size_t off = 0;
  auto alloc = [&](size_t bytes) -> void* {
    void* p = ws + off;
    off = (off + bytes + 255) & ~(size_t)255;
    return p;
  };
  float*    xbar      = (float*)alloc((size_t)NN * CIN * VV * 4);
  float*    Agg       = (float*)alloc((size_t)NN * JJ * VV * VV * 4);
  float*    pre_scale = (float*)alloc(JJ * 4);
  float*    pre_shift = (float*)alloc(JJ * 4);
  float*    sbn       = (float*)alloc(COUT * 4);
  float*    sdn       = (float*)alloc(COUT * 4);
  float*    fshift    = (float*)alloc(COUT * 4);
  uint16_t* x_sw      = (uint16_t*)alloc((size_t)NN * 4 * 200 * 512 * 2);      // 52.4 MB
  uint16_t* pre_sw    = (uint16_t*)alloc((size_t)NN * JJ * 8 * 512 * 2);       // 50.3 MB
  uint16_t* y_sw      = (uint16_t*)alloc((size_t)NN * 3 * 200 * 512 * 2);      // 39.3 MB
  uint16_t* w_pre_sw  = (uint16_t*)alloc((size_t)6 * 4 * 512 * 2);
  uint16_t* w_down_sw = (uint16_t*)alloc((size_t)16 * 4 * 512 * 2);
  uint16_t* w_post_sw = (uint16_t*)alloc((size_t)16 * 3 * 512 * 2);
  (void)ws_size; (void)n_in; (void)in_sizes; (void)out_size;

  // 0) fold BN, swizzle weights; zero pre_sw pad lanes
  const int coeff_work = JJ*CIN + COUT*CIN + COUT*JJ + JJ + COUT;
  k_coeffs<<<(coeff_work + 255) / 256, 256, 0, stream>>>(
      w_pre, w_down, w_post, b_pre, g_pre, bb_pre, m_pre, v_pre,
      b_post, b_down, g_down, bb_down, m_down, v_down, g_bn, bb_bn, m_bn, v_bn,
      w_pre_sw, w_down_sw, w_post_sw, pre_scale, pre_shift, sbn, sdn, fshift);
  const int pre_n4 = (int)(((size_t)NN * JJ * 8 * 512 * 2) / 16);
  k_zero<<<(pre_n4 + 255) / 256, 256, 0, stream>>>((u32x4*)pre_sw, pre_n4);

  // 1) x -> swizzled bf16 B-tiles + time mean
  k_reduce<<<NN * 4, 256, 0, stream>>>(x, x_sw, xbar);

  // 2) adjacency construction (writes get_graph + Agg)
  k_graph<<<NN * KK, 256, 0, stream>>>(xbar, w1, b1, w2, b2, A, alpha, beta, Agg, gg);

  // 3) pre branch GEMM: 64 * (96x128 @ 128x3200), fused BN+ReLU, 4 tiles/wave
  k_pre<<<(NN * 6 * 50) / 8, 256, 0, stream>>>(x_sw, w_pre_sw, pre_scale, pre_shift, pre_sw);

  // 4) per-channel graph aggregation
  k_agg<<<NN * JJ, 128, 0, stream>>>(pre_sw, Agg, y_sw);

  // 5) fused post-conv + down-conv residual + BN + ReLU, 2 tiles/wave, dual GEMM
  k_final<<<(NN * 16 * 100) / 8, 256, 0, stream>>>(
      y_sw, x_sw, w_post_sw, w_down_sw, sbn, sdn, fshift, out);
}
